// KuramotoLayer_80298708566056
// MI455X (gfx1250) — matrix operand order repaired and checked by reading, not compiled, not run on hardware
//
#include <hip/hip_runtime.h>
#include <math.h>

#define N_OSC   256
#define BATCH   1024
#define DT      0.1f
#define STEPS   10
#define TILE_M  16
#define KST     264      // padded LDS row stride (elements); 264*2B = 528B -> 4-bank shift per row
#define THREADS 256

typedef _Float16 v16h __attribute__((ext_vector_type(16)));
typedef _Float16 v8h  __attribute__((ext_vector_type(8)));
typedef float    v8f  __attribute__((ext_vector_type(8)));

__global__ __launch_bounds__(THREADS, 1)
void kuramoto_wmma_kernel(const float* __restrict__ theta_init,
                          const float* __restrict__ Kmat,
                          const float* __restrict__ omega,
                          const float* __restrict__ Kglob,
                          float* __restrict__ out_theta,
                          float* __restrict__ out_coh)
{
    // LDS: 135168 + 16896 + 8448 + 8448 + 1024 = 169984 B (of 320KB WGP LDS)
    __shared__ _Float16 Klds[N_OSC * KST];   // K as f16, [i][j] == [j][i] (symmetric)
    __shared__ float    th  [TILE_M * KST];  // current theta tile (f32)
    __shared__ _Float16 sS  [TILE_M * KST];  // sin(theta) tile (f16, WMMA A operand)
    __shared__ _Float16 sC  [TILE_M * KST];  // cos(theta) tile (f16, WMMA A operand)
    __shared__ float    som [N_OSC];         // omega

    const int tid  = threadIdx.x;
    const int lane = tid & 31;               // wave32
    const int wid  = tid >> 5;               // 8 waves
    const int bb   = blockIdx.x * TILE_M;    // batch-row base

    // ---- stage K (f32 -> f16) ----
    for (int idx = tid; idx < N_OSC * N_OSC; idx += THREADS) {
        int i = idx >> 8, j = idx & 255;
        Klds[i * KST + j] = (_Float16)Kmat[idx];
    }
    for (int j = tid; j < N_OSC; j += THREADS) som[j] = omega[j];

    // ---- stage theta tile + initial sin/cos ----
    for (int idx = tid; idx < TILE_M * N_OSC; idx += THREADS) {
        int m = idx >> 8, j = idx & 255;     // j == tid (coalesced global read)
        float t = theta_init[(bb + m) * N_OSC + j];
        float s, c; sincosf(t, &s, &c);
        th[m * KST + j] = t;
        sS[m * KST + j] = (_Float16)s;
        sC[m * KST + j] = (_Float16)c;
    }
    const float scale = Kglob[0] * (1.0f / N_OSC);
    __syncthreads();

    const int m_a  = lane & 15;   // A-matrix row for this lane
    const int grp  = lane >> 4;   // lane half (selects K sub-range per ISA layout)
    const int ncol = lane & 15;   // B/D column within tile
    const int nt0  = wid * 2;     // this wave's two N-tiles
    const int nt1  = nt0 + 1;

    for (int step = 0; step < STEPS; ++step) {
        v8f accS0 = {}; v8f accC0 = {};   // S*K, C*K for tile nt0
        v8f accS1 = {}; v8f accC1 = {};   // ... for tile nt1

        // ---- phase A: D = A(16x32) x B(32x16) over K-dim in chunks of 32 ----
        #pragma unroll
        for (int kb = 0; kb < N_OSC; kb += 32) {
            // A operand: 16-bit 16x32 layout -> elements 0..15: k = kb+grp*16+e
            //            (contiguous, matching B)
            const int ka0 = kb + grp * 16;
            const int ka1 = ka0 + 8;
            v8h as0 = *(const v8h*)&sS[m_a * KST + ka0];
            v8h as1 = *(const v8h*)&sS[m_a * KST + ka1];
            v8h ac0 = *(const v8h*)&sC[m_a * KST + ka0];
            v8h ac1 = *(const v8h*)&sC[m_a * KST + ka1];
            v16h aS, aC;
            #pragma unroll
            for (int i = 0; i < 8; ++i) {
                aS[i] = as0[i]; aS[8 + i] = as1[i];
                aC[i] = ac0[i]; aC[8 + i] = ac1[i];
            }
            // B operand: 16-bit 32x16 layout -> lane holds column n = nt*16+ncol,
            // rows k = kb + grp*16 .. +15 (contiguous; K symmetric so [n][k] works)
            const int kbrow = kb + grp * 16;
            const _Float16* bp0 = &Klds[(nt0 * 16 + ncol) * KST + kbrow];
            const _Float16* bp1 = &Klds[(nt1 * 16 + ncol) * KST + kbrow];
            v8h b0lo = *(const v8h*)bp0;
            v8h b0hi = *(const v8h*)(bp0 + 8);
            v8h b1lo = *(const v8h*)bp1;
            v8h b1hi = *(const v8h*)(bp1 + 8);
            v16h b0, b1;
            #pragma unroll
            for (int i = 0; i < 8; ++i) {
                b0[i] = b0lo[i]; b0[8 + i] = b0hi[i];
                b1[i] = b1lo[i]; b1[8 + i] = b1hi[i];
            }
            accS0 = __builtin_amdgcn_wmma_f32_16x16x32_f16(false, aS, false, b0, (short)0, accS0, false, false);
            accC0 = __builtin_amdgcn_wmma_f32_16x16x32_f16(false, aC, false, b0, (short)0, accC0, false, false);
            accS1 = __builtin_amdgcn_wmma_f32_16x16x32_f16(false, aS, false, b1, (short)0, accS1, false, false);
            accC1 = __builtin_amdgcn_wmma_f32_16x16x32_f16(false, aC, false, b1, (short)0, accC1, false, false);
        }
        __syncthreads();   // all waves done reading sS/sC before owners rewrite

        // ---- phase B: elementwise update of this wave's two tiles ----
        // D layout: VGPR v -> M = v + 8*grp, N = ncol (+ tile base)
        #pragma unroll
        for (int v = 0; v < 8; ++v) {
            const int M = v + 8 * grp;
            {
                const int N = nt0 * 16 + ncol;
                float t = th[M * KST + N];
                float s = (float)sS[M * KST + N];
                float c = (float)sC[M * KST + N];
                float coupling = c * accS0[v] - s * accC0[v];
                float tn = t + DT * (som[N] + scale * coupling);
                float s2, c2; sincosf(tn, &s2, &c2);
                th[M * KST + N] = atan2f(s2, c2);       // wrap to (-pi, pi]
                sS[M * KST + N] = (_Float16)s2;         // sin/cos invariant under wrap
                sC[M * KST + N] = (_Float16)c2;
            }
            {
                const int N = nt1 * 16 + ncol;
                float t = th[M * KST + N];
                float s = (float)sS[M * KST + N];
                float c = (float)sC[M * KST + N];
                float coupling = c * accS1[v] - s * accC1[v];
                float tn = t + DT * (som[N] + scale * coupling);
                float s2, c2; sincosf(tn, &s2, &c2);
                th[M * KST + N] = atan2f(s2, c2);
                sS[M * KST + N] = (_Float16)s2;
                sC[M * KST + N] = (_Float16)c2;
            }
        }
        __syncthreads();
    }

    // ---- write theta ----
    for (int idx = tid; idx < TILE_M * N_OSC; idx += THREADS) {
        int m = idx >> 8, j = idx & 255;
        out_theta[(bb + m) * N_OSC + j] = th[m * KST + j];
    }
    // ---- coherence per batch row ----
    if (tid < TILE_M) {
        float sums = 0.0f, sumc = 0.0f;
        for (int j = 0; j < N_OSC; ++j) {
            float t = th[tid * KST + j];
            float s, c; sincosf(t, &s, &c);
            sums += s; sumc += c;
        }
        sums *= (1.0f / N_OSC);
        sumc *= (1.0f / N_OSC);
        out_coh[bb + tid] = sqrtf(sums * sums + sumc * sumc);
    }
}

extern "C" void kernel_launch(void* const* d_in, const int* in_sizes, int n_in,
                              void* d_out, int out_size, void* d_ws, size_t ws_size,
                              hipStream_t stream) {
    const float* theta_init = (const float*)d_in[0];   // (1024, 256)
    const float* Kmat       = (const float*)d_in[1];   // (256, 256)
    const float* omega      = (const float*)d_in[2];   // (256,)
    const float* Kglob      = (const float*)d_in[3];   // scalar
    float* out_theta = (float*)d_out;                  // 1024*256
    float* out_coh   = out_theta + BATCH * N_OSC;      // 1024

    dim3 grid(BATCH / TILE_M);   // 64 workgroups, 8 waves each
    kuramoto_wmma_kernel<<<grid, THREADS, 0, stream>>>(
        theta_init, Kmat, omega, Kglob, out_theta, out_coh);
}
